// QuantizedLinear_86346022518964
// MI455X (gfx1250) — matrix operand order repaired and checked by reading, not compiled, not run on hardware
//
#include <hip/hip_runtime.h>

// ---------------------------------------------------------------------------
// QuantizedLinear for MI455X (gfx1250):
//   out = x @ quantize(weight).T,  x:(16384 x 4096) f32, w:(4096 x 4096) f32
// GEMM M=16384 N=4096 K=4096 -> 549 GFLOP vs 576MB unique traffic: compute
// bound, so drive v_wmma_f32_16x16x32_bf16.
//  * weight quantizer: qw = n/(9*alpha), n = clamp(round(9*alpha*w),-2,2).
//    n is an exact bf16 (zero low mantissa bits -> byte-perm pack, no
//    rounding math); scale 1/(9*alpha) folded into the epilogue.
//  * activations: truncation split x = hi + lo (bf16 each), two WMMA passes
//    into the same f32 accumulator -> ~2^-16 relative error (near fp32) at
//    3 VALU/element staging cost, hidden under the WMMA shadow.
//  * double-buffered, bank-conflict-padded LDS; one barrier per K-chunk;
//    global float4 prefetch overlapped with the WMMA burst.
// ---------------------------------------------------------------------------

typedef __bf16 bf16_t;
typedef __attribute__((ext_vector_type(16))) bf16_t v16bf;
typedef __attribute__((ext_vector_type(8)))  float  v8f;

#define TILE_M 128
#define TILE_N 128
#define TILE_K 32
#define K_DIM  4096
#define N_DIM  4096
#define LDSW   20          // row stride in uints: 16 data + 4 pad (bank-safe, 16B-aligned)

// v_perm_b32 selector: dst = {S1.b2, S1.b3, S0.b2, S0.b3}  (sel bytes 0-3 pick
// from S1, 4-7 from S0). With S0 = hi(second elem), S1 = hi(first elem) this
// packs two bf16 (top halves of two f32) into one dword.
#define PACK_HI2(second, first) __builtin_amdgcn_perm((second), (first), 0x07060302u)

union FragU {              // one 16x32 bf16 operand fragment: 8 VGPRs per lane
  v16bf v;
  uint4 q[2];
};

__global__ __launch_bounds__(256)
void QuantizedLinear_wmma_bf16x2(const float* __restrict__ x,
                                 const float* __restrict__ w,
                                 const float* __restrict__ alphap,
                                 float* __restrict__ out) {
  // Double-buffered bf16-packed LDS tiles: 3 * 2 * 128 * 20 * 4B = 60 KB.
  __shared__ unsigned Ah[2][TILE_M][LDSW];   // x hi parts
  __shared__ unsigned Al[2][TILE_M][LDSW];   // x lo parts
  __shared__ unsigned Bq[2][TILE_N][LDSW];   // quantized weight ints (exact)

  const int tid  = threadIdx.x;
  const int lane = tid & 31;
  const int wave = tid >> 5;           // 8 waves
  const int wm   = wave >> 2;          // 0..1  (64 rows each)
  const int wn   = wave & 3;           // 0..3  (32 cols each)
  const int half = lane >> 4;          // 0..1
  const int l16  = lane & 15;

  const float alpha = alphap[0];
  const float qs    = 9.0f * alpha;          // w -> signed level index
  const float sc    = 1.0f / (9.0f * alpha); // epilogue scale (qw = n * sc)

  const int bn = blockIdx.x;           // N tile
  const int bm = blockIdx.y;           // M tile

  const float* xbase = x + (size_t)bm * TILE_M * K_DIM;
  const float* wbase = w + (size_t)bn * TILE_N * K_DIM;

  v8f acc[4][2];
#pragma unroll
  for (int fm = 0; fm < 4; ++fm)
#pragma unroll
    for (int fn = 0; fn < 2; ++fn)
#pragma unroll
      for (int j = 0; j < 8; ++j) acc[fm][fn][j] = 0.0f;

  // 256 threads move one 128x32 f32 tile per matrix as 4 float4 each.
  float4 ra[4], rb[4];
  auto load_tiles = [&](int kk) {
#pragma unroll
    for (int j = 0; j < 4; ++j) {
      const int idx = tid + 256 * j;       // 0..1023
      const int row = idx >> 3;            // 0..127
      const int c4  = idx & 7;             // float4 slot within 32-wide row
      ra[j] = *reinterpret_cast<const float4*>(xbase + (size_t)row * K_DIM + kk + c4 * 4);
      rb[j] = *reinterpret_cast<const float4*>(wbase + (size_t)row * K_DIM + kk + c4 * 4);
    }
  };

  // Stage registers -> LDS[p] with fused split / quantize (low VALU count).
  auto stage = [&](int p) {
#pragma unroll
    for (int j = 0; j < 4; ++j) {
      const int idx = tid + 256 * j;
      const int row = idx >> 3;
      const int c4  = idx & 7;

      // ---- x: truncation split. hi = top16(a); lo = a - hi (exact). ----
      const float4 a = ra[j];
      const unsigned hx = __float_as_uint(a.x) & 0xffff0000u;
      const unsigned hy = __float_as_uint(a.y) & 0xffff0000u;
      const unsigned hz = __float_as_uint(a.z) & 0xffff0000u;
      const unsigned hw = __float_as_uint(a.w) & 0xffff0000u;
      const unsigned lx = __float_as_uint(a.x - __uint_as_float(hx));
      const unsigned ly = __float_as_uint(a.y - __uint_as_float(hy));
      const unsigned lz = __float_as_uint(a.z - __uint_as_float(hz));
      const unsigned lw = __float_as_uint(a.w - __uint_as_float(hw));
      uint2 hp, lp;
      hp.x = PACK_HI2(hy, hx);
      hp.y = PACK_HI2(hw, hz);
      lp.x = PACK_HI2(ly, lx);      // truncate lo to bf16 (err <= 2^-16 |a|)
      lp.y = PACK_HI2(lw, lz);
      *reinterpret_cast<uint2*>(&Ah[p][row][c4 * 2]) = hp;
      *reinterpret_cast<uint2*>(&Al[p][row][c4 * 2]) = lp;

      // ---- weight: n = clamp(round(9*alpha*w), -2, 2); exact in bf16. ----
      const float4 b = rb[j];
      const float q0 = fminf(2.0f, fmaxf(-2.0f, rintf(qs * b.x)));
      const float q1 = fminf(2.0f, fmaxf(-2.0f, rintf(qs * b.y)));
      const float q2 = fminf(2.0f, fmaxf(-2.0f, rintf(qs * b.z)));
      const float q3 = fminf(2.0f, fmaxf(-2.0f, rintf(qs * b.w)));
      uint2 qp;
      qp.x = PACK_HI2(__float_as_uint(q1), __float_as_uint(q0));
      qp.y = PACK_HI2(__float_as_uint(q3), __float_as_uint(q2));
      *reinterpret_cast<uint2*>(&Bq[p][row][c4 * 2]) = qp;
    }
  };

  load_tiles(0);

  int p = 0;
  for (int kk = 0; kk < K_DIM; kk += TILE_K, p ^= 1) {
    stage(p);
    __syncthreads();   // single barrier per chunk: writes went to buffer p,
                       // buffer p was last read two chunks ago, before the previous barrier

    // prefetch next K-chunk while the WMMA burst runs
    if (kk + TILE_K < K_DIM) load_tiles(kk + TILE_K);

    // ---- fragment loads (ds_load_b128, padded stride -> conflict-free) ----
    // A layout (16-bit 16x32): lane(half=0): K={0..7,16..23}; half=1: {8..15,24..31}
    FragU ah[4], al[4];
#pragma unroll
    for (int fm = 0; fm < 4; ++fm) {
      const int m0 = wm * 64 + fm * 16 + l16;
      const uint4* ph = reinterpret_cast<const uint4*>(&Ah[p][m0][half * 4]);
      ah[fm].q[0] = ph[0];
      ah[fm].q[1] = ph[2];                  // +8 uints = K+16
      const uint4* pl = reinterpret_cast<const uint4*>(&Al[p][m0][half * 4]);
      al[fm].q[0] = pl[0];
      al[fm].q[1] = pl[2];
    }
    // B layout (16-bit 32x16): lane n = l16, half=0: K={0..7,16..23}; half=1: {8..15,24..31}
    FragU bfr[2];
#pragma unroll
    for (int fn = 0; fn < 2; ++fn) {
      const int n0 = wn * 32 + fn * 16 + l16;
      const uint4* pb = reinterpret_cast<const uint4*>(&Bq[p][n0][half * 4]);
      bfr[fn].q[0] = pb[0];
      bfr[fn].q[1] = pb[2];
    }

    // ---- 16 WMMAs: hi pass then lo pass (same-acc reuse distance = 8) ----
#pragma unroll
    for (int fm = 0; fm < 4; ++fm)
#pragma unroll
      for (int fn = 0; fn < 2; ++fn)
        acc[fm][fn] = __builtin_amdgcn_wmma_f32_16x16x32_bf16(
            false, ah[fm].v, false, bfr[fn].v, (short)0, acc[fm][fn], false, false);
#pragma unroll
    for (int fm = 0; fm < 4; ++fm)
#pragma unroll
      for (int fn = 0; fn < 2; ++fn)
        acc[fm][fn] = __builtin_amdgcn_wmma_f32_16x16x32_bf16(
            false, al[fm].v, false, bfr[fn].v, (short)0, acc[fm][fn], false, false);
  }

  // ---- epilogue: scale by 1/(9*alpha), write f32 ----
  // C/D layout: VGPR j, lanes0-15 -> M=j, lanes16-31 -> M=8+j; N = l16
#pragma unroll
  for (int fm = 0; fm < 4; ++fm) {
    const int row0 = bm * TILE_M + wm * 64 + fm * 16 + half * 8;
#pragma unroll
    for (int fn = 0; fn < 2; ++fn) {
      const int col = bn * TILE_N + wn * 32 + fn * 16 + l16;
#pragma unroll
      for (int j = 0; j < 8; ++j) {
        out[(size_t)(row0 + j) * N_DIM + col] = acc[fm][fn][j] * sc;
      }
    }
  }
}

extern "C" void kernel_launch(void* const* d_in, const int* in_sizes, int n_in,
                              void* d_out, int out_size, void* d_ws, size_t ws_size,
                              hipStream_t stream) {
  const float* x     = (const float*)d_in[0];  // (8,2048,4096) f32
  const float* w     = (const float*)d_in[1];  // (4096,4096)  f32
  const float* alpha = (const float*)d_in[2];  // scalar
  // d_in[3] (voltage_levels) is the fixed uniform grid {3.5..5.5 step 0.5};
  // nearest-level selection is computed in closed form in the kernel.
  float* out = (float*)d_out;

  const int M = in_sizes[0] / K_DIM;           // 16384, divisible by TILE_M
  dim3 grid(N_DIM / TILE_N, M / TILE_M);       // (32, 128)
  QuantizedLinear_wmma_bf16x2<<<grid, 256, 0, stream>>>(x, w, alpha, out);
}
